// CoxPH_12618613916173
// MI455X (gfx1250) — matrix operand order repaired and checked by reading, not compiled, not run on hardware
//
#include <hip/hip_runtime.h>
#include <hip/hip_bf16.h>
#include <math.h>

#define NN 8192

typedef __attribute__((ext_vector_type(16))) _Float16 v16h;
typedef __attribute__((ext_vector_type(8)))  _Float16 v8h;
typedef __attribute__((ext_vector_type(8)))  float    v8f;
typedef __attribute__((ext_vector_type(4)))  float    v4f;

// ---------------------------------------------------------------------------
// Kernel 1: e = exp(hazard); split into f16 hi/lo so that e_hi + e_lo ~= e
// with ~f32 precision (used by the two-pass f16 WMMA accumulation).
// ---------------------------------------------------------------------------
__global__ __launch_bounds__(256) void cox_prep_kernel(
    const float* __restrict__ hazard,
    _Float16* __restrict__ e_hi,
    _Float16* __restrict__ e_lo) {
    int i = blockIdx.x * blockDim.x + threadIdx.x;
    if (i < NN) {
        float e = expf(hazard[i]);
        _Float16 hi = (_Float16)e;
        e_hi[i] = hi;
        e_lo[i] = (_Float16)(e - (float)hi);
    }
}

// ---------------------------------------------------------------------------
// Kernel 2: risk_sum[i] = sum_j (time[j] >= time[i]) * exp(hazard[j])
// One block = 8 waves handles a 16-row tile. Each wave builds the 0/1 mask in
// the B operand (lane owns column n = lane%16, K = 8*(lane/16)+h (h<8) or 16+8*(lane/16)+h-8 (h>=8)) and the
// replicated e_hi/e_lo values in the A operand (layout per CDNA5 ISA 7.12.2),
// then accumulates with v_wmma_f32_16x16x32_f16. Wave partials are combined
// through LDS in a fixed order -> bitwise deterministic (no float atomics).
// ---------------------------------------------------------------------------
__global__ __launch_bounds__(256) void cox_risk_kernel(
    const float* __restrict__ time,
    const _Float16* __restrict__ e_hi,
    const _Float16* __restrict__ e_lo,
    float* __restrict__ risk) {
    __shared__ float part[8 * 16];

    const int lane = threadIdx.x & 31;   // wave32
    const int wv   = threadIdx.x >> 5;   // 8 waves / block
    const int half = lane >> 4;          // which 16-lane half of the wave
    const int ncol = lane & 15;          // output row handled by this lane
    const int i0   = blockIdx.x * 16;    // row tile base

    const float trow = time[i0 + ncol];

    v8f c = {};

    // 8192 columns = 256 chunks of K=32; wave wv takes chunks wv, wv+8, ...
    for (int ch = wv; ch < NN / 32; ch += 8) {
        const int j0 = ch * 32;

        // ---- A operand: e values, replicated over M ----
        // lane<16: K = 0..7 and 16..23 ; lane>=16: K = 8..15 and 24..31
        const v8h a0h = *reinterpret_cast<const v8h*>(e_hi + j0 + 8 * half);
        const v8h a1h = *reinterpret_cast<const v8h*>(e_hi + j0 + 16 + 8 * half);
        const v8h a0l = *reinterpret_cast<const v8h*>(e_lo + j0 + 8 * half);
        const v8h a1l = *reinterpret_cast<const v8h*>(e_lo + j0 + 16 + 8 * half);
        v16h ahi = __builtin_shufflevector(a0h, a1h, 0,1,2,3,4,5,6,7,8,9,10,11,12,13,14,15);
        v16h alo = __builtin_shufflevector(a0l, a1l, 0,1,2,3,4,5,6,7,8,9,10,11,12,13,14,15);

        // ---- B operand: risk-set mask, lane holds K = 8*half + h (h<8), 16 + 8*half + (h-8) (h>=8) ----
        const float* tp = time + j0 + 8 * half;
        v16h b;
        #pragma unroll
        for (int q = 0; q < 4; ++q) {
            v4f t4 = *reinterpret_cast<const v4f*>(tp + 4 * q + 8 * (q >> 1));
            #pragma unroll
            for (int r = 0; r < 4; ++r) {
                b[4 * q + r] = (t4[r] >= trow) ? (_Float16)1.0f : (_Float16)0.0f;
            }
        }

        // Two WMMAs against the same mask: hi + lo split restores f32 accuracy.
        c = __builtin_amdgcn_wmma_f32_16x16x32_f16(false, ahi, false, b,
                                                   (short)0, c, false, false);
        c = __builtin_amdgcn_wmma_f32_16x16x32_f16(false, alo, false, b,
                                                   (short)0, c, false, false);
    }

    // D[m][n] is identical across m (A replicated over M); lanes 0..15 hold
    // the partial for row i0 + lane in c[0].
    if (lane < 16) part[wv * 16 + lane] = c[0];
    __syncthreads();

    if (threadIdx.x < 16) {
        float s = 0.0f;
        #pragma unroll
        for (int w = 0; w < 8; ++w) s += part[w * 16 + threadIdx.x];
        risk[i0 + threadIdx.x] = s;
    }
}

// ---------------------------------------------------------------------------
// Kernel 3: loss = -mean((hazard - log(risk_sum)) * censor)
// Single block, fixed-order tree reduction -> deterministic.
// ---------------------------------------------------------------------------
__global__ __launch_bounds__(256) void cox_loss_kernel(
    const float* __restrict__ hazard,
    const float* __restrict__ censor,
    const float* __restrict__ risk,
    float* __restrict__ out) {
    __shared__ float red[256];
    float acc = 0.0f;
    for (int i = threadIdx.x; i < NN; i += 256) {
        acc += (hazard[i] - logf(risk[i])) * censor[i];
    }
    red[threadIdx.x] = acc;
    __syncthreads();
    #pragma unroll
    for (int s = 128; s > 0; s >>= 1) {
        if (threadIdx.x < s) red[threadIdx.x] += red[threadIdx.x + s];
        __syncthreads();
    }
    if (threadIdx.x == 0) out[0] = -red[0] / (float)NN;
}

// ---------------------------------------------------------------------------
extern "C" void kernel_launch(void* const* d_in, const int* in_sizes, int n_in,
                              void* d_out, int out_size, void* d_ws, size_t ws_size,
                              hipStream_t stream) {
    (void)in_sizes; (void)n_in; (void)out_size; (void)ws_size;

    const float* hazard = (const float*)d_in[0];
    const float* time_v = (const float*)d_in[1];
    const float* censor = (const float*)d_in[2];
    float* out = (float*)d_out;

    // Workspace layout (64 KB): risk f32[8192] | e_hi f16[8192] | e_lo f16[8192]
    char* ws = (char*)d_ws;
    float*    risk = (float*)ws;                       // 32768 bytes
    _Float16* e_hi = (_Float16*)(ws + 32768);          // 16384 bytes
    _Float16* e_lo = (_Float16*)(ws + 32768 + 16384);  // 16384 bytes

    cox_prep_kernel<<<NN / 256, 256, 0, stream>>>(hazard, e_hi, e_lo);
    cox_risk_kernel<<<NN / 16, 256, 0, stream>>>(time_v, e_hi, e_lo, risk);
    cox_loss_kernel<<<1, 256, 0, stream>>>(hazard, censor, risk, out);
}
